// LSTMencoder_61314953117813
// MI455X (gfx1250) — matrix operand order repaired and checked by reading, not compiled, not run on hardware
//
#include <hip/hip_runtime.h>
#include <cstdint>

typedef __attribute__((ext_vector_type(16))) __bf16 v16bf;
typedef __attribute__((ext_vector_type(8)))  __bf16 v8bf;
typedef __attribute__((ext_vector_type(8)))  float  v8f;
typedef int v4i_ __attribute__((vector_size(4 * sizeof(int))));

#define DEV __device__ __forceinline__

#if defined(__has_builtin)
#if __has_builtin(__builtin_amdgcn_global_load_async_to_lds_b128) && \
    __has_builtin(__builtin_amdgcn_s_wait_asynccnt)
#define ATHENA_ASYNC_LDS 1
#endif
#endif

DEV unsigned short f32_bf16(float f) {
  union { float f; unsigned u; } x; x.f = f;
  unsigned u = x.u;
  return (unsigned short)((u + 0x7FFFu + ((u >> 16) & 1u)) >> 16);
}
DEV float bf16_f32(unsigned short h) {
  union { unsigned u; float f; } x; x.u = ((unsigned)h) << 16; return x.f;
}
DEV float sigm(float x) { return 1.0f / (1.0f + __expf(-x)); }

// ---- WMMA fragment helpers (wave32, 16x16x32 bf16) ----------------------
// A (16x32, row-major source): lane L holds row (L&15);
//   elems 0..15 -> K = k0 + (L>>4)*16 + j
DEV v16bf load_a_row(const unsigned short* rowp, int k0, int lane) {
  const unsigned short* p = rowp + k0 + ((lane >> 4) << 4);
  union { v16bf v; v8bf h[2]; } u;
  u.h[0] = *(const v8bf*)(p);
  u.h[1] = *(const v8bf*)(p + 8);
  return u.v;
}
// B pre-swizzled: element (k,n) at ((nTile*kTiles + kTile)*32 + L)*16 + j
// with L = ((k>>4)&1)*16 | (n&15), j = k&15 -> 32 contiguous bytes per lane.
DEV v16bf load_b_sw(const unsigned short* bsw, int kTiles, int nTile, int kT, int lane) {
  const unsigned short* p = bsw + (((size_t)(nTile * kTiles + kT) * 32 + lane) << 4);
  union { v16bf v; v8bf h[2]; } u;
  u.h[0] = *(const v8bf*)(p);
  u.h[1] = *(const v8bf*)(p + 8);
  return u.v;
}
DEV v8f wmma_bf16(v16bf a, v16bf b, v8f c) {
  return __builtin_amdgcn_wmma_f32_16x16x32_bf16(false, a, false, b, (short)0, c, false, false);
}

// ---- prep kernels --------------------------------------------------------
__global__ void k_f32_to_bf16(const float* __restrict__ src,
                              unsigned short* __restrict__ dst, long long n) {
  long long i = (long long)blockIdx.x * blockDim.x + threadIdx.x;
  long long s = (long long)gridDim.x * blockDim.x;
  for (; i < n; i += s) dst[i] = f32_bf16(src[i]);
}

// Convert KxN f32 weight to bf16, optional gate-interleave (col' = 4u+g),
// store in WMMA-B swizzled layout.
__global__ void k_prep_w(const float* __restrict__ src,
                         unsigned short* __restrict__ dst,
                         int K, int N, int permute) {
  long long tot = (long long)K * N;
  long long i = (long long)blockIdx.x * blockDim.x + threadIdx.x;
  long long s = (long long)gridDim.x * blockDim.x;
  int kTiles = K >> 5;
  for (; i < tot; i += s) {
    int n = (int)(i % N);
    int k = (int)(i / N);
    int nsrc = permute ? ((n & 3) * (N >> 2) + (n >> 2)) : n;
    float v = src[(size_t)k * N + nsrc];
    int lane = (((k >> 4) & 1) << 4) | (n & 15);
    size_t idx = (((size_t)(n >> 4) * kTiles + (k >> 5)) * 32 + lane) * 16 + (k & 15);
    dst[idx] = f32_bf16(v);
  }
}

__global__ void k_prep_b(const float* __restrict__ src, float* __restrict__ dst, int N) {
  int n = blockIdx.x * blockDim.x + threadIdx.x;
  if (n < N) dst[n] = src[(n & 3) * (N >> 2) + (n >> 2)];
}

// ---- phase 1: xz = gather(A)[.] @ Bsw + bias  (M x 2048, K = 512) --------
__global__ __launch_bounds__(256) void k_gemm_xz(
    const int* __restrict__ tokens,          // row gather indices or nullptr
    const unsigned short* __restrict__ A,    // [rows][512] bf16
    const unsigned short* __restrict__ Bsw,  // swizzled 512x2048 bf16
    const float* __restrict__ bias,          // [2048] permuted
    unsigned short* __restrict__ Cout)       // [M][2048] bf16
{
  const int K = 512, N = 2048, kTiles = K >> 5;
  int mBase = blockIdx.x * 64;
  __shared__ int tok[64];
  if (threadIdx.x < 64) {
    int m = mBase + threadIdx.x;
    tok[threadIdx.x] = tokens ? tokens[m] : m;
  }
  __syncthreads();
  int wave = threadIdx.x >> 5, lane = threadIdx.x & 31;
  int nTile = blockIdx.y * 8 + wave;

  v8f acc[4];
  v8f zero = {};
  #pragma unroll
  for (int mT = 0; mT < 4; ++mT) acc[mT] = zero;

  for (int kT = 0; kT < kTiles; ++kT) {
    v16bf b = load_b_sw(Bsw, kTiles, nTile, kT, lane);
    int k0 = kT << 5;
    #pragma unroll
    for (int mT = 0; mT < 4; ++mT) {
      const unsigned short* rowp = A + (size_t)tok[mT * 16 + (lane & 15)] * K;
      v16bf a = load_a_row(rowp, k0, lane);
      acc[mT] = wmma_bf16(a, b, acc[mT]);
    }
  }
  int n = (nTile << 4) + (lane & 15);
  float bv = bias[n];
  #pragma unroll
  for (int mT = 0; mT < 4; ++mT)
    #pragma unroll
    for (int v = 0; v < 8; ++v) {
      int m = mBase + mT * 16 + v + ((lane >> 4) << 3);
      Cout[(size_t)m * N + n] = f32_bf16(acc[mT][v] + bv);
    }
}

// ---- phase 2: sequential LSTM scan over T steps --------------------------
// 16 persistent workgroups, each owns 128 gate-interleaved columns
// (32 hidden units). U slice (128KB) staged to LDS with ASYNC-to-LDS DMA;
// h ping-pongs in global; device-scope spin barrier per step; cell state c
// lives in VGPRs for the entire 1024-step scan.
__global__ __launch_bounds__(256) void k_lstm_scan(
    const unsigned short* __restrict__ xz,    // [64*T][2048] bf16 (interleaved)
    const unsigned short* __restrict__ Usw,   // swizzled 512x2048 bf16
    unsigned short* __restrict__ hbuf,        // [2][64][512] bf16 ping-pong
    unsigned short* __restrict__ h1out,       // [64][T][512] bf16 or nullptr
    unsigned int* __restrict__ bar,
    int T)
{
  const int K = 512, N = 2048, kTiles = 16, NWG = 16;
  extern __shared__ unsigned short Ulds[];    // 65536 elems = 128 KB
  int wg = blockIdx.x, tid = threadIdx.x;

  // ---- stage this WG's contiguous 128KB U slice (nTiles [wg*8, wg*8+8)) --
#ifdef ATHENA_ASYNC_LDS
  {
    typedef __attribute__((address_space(1))) v4i_ gvec;
    typedef __attribute__((address_space(3))) v4i_ lvec;
    gvec* s = (gvec*)(Usw + (size_t)wg * 65536);
    lvec* d = (lvec*)Ulds;
    // 8192 x 16B chunks, 32 async b128 transfers per thread, no VGPR staging
    #pragma unroll 4
    for (int i = tid; i < 8192; i += 256)
      __builtin_amdgcn_global_load_async_to_lds_b128(s + i, d + i, 0, 0);
    __builtin_amdgcn_s_wait_asynccnt(0);
  }
#else
  {
    const uint4* s = (const uint4*)(Usw + (size_t)wg * 65536);
    uint4* d = (uint4*)Ulds;
    for (int i = tid; i < 8192; i += 256) d[i] = s[i];
  }
#endif
  __syncthreads();

  int wave = tid >> 5, lane = tid & 31;
  int nTloc = wave;                              // one 16-col tile per wave
  int n = wg * 128 + (nTloc << 4) + (lane & 15); // global z column
  int group = (lane & 15) >> 2;                  // hidden unit within tile
  int u = ((wg * 128 + (nTloc << 4)) >> 2) + group;
  int baseLane = lane & ~3;

  float c[4][8];
  #pragma unroll
  for (int mT = 0; mT < 4; ++mT)
    #pragma unroll
    for (int v = 0; v < 8; ++v) c[mT][v] = 0.0f;

  for (int t = 0; t < T; ++t) {
    const unsigned short* hb = hbuf + (size_t)(t & 1) * (64 * 512);
    unsigned short*       hw = hbuf + (size_t)((t + 1) & 1) * (64 * 512);

    #pragma unroll
    for (int mT = 0; mT < 4; ++mT) {
      v8f acc;
      #pragma unroll
      for (int v = 0; v < 8; ++v) {        // z := xz[:, t, :]
        int m = mT * 16 + v + ((lane >> 4) << 3);
        acc[v] = bf16_f32(xz[((size_t)m * T + t) * N + n]);
      }
      // prefetch next step's xz line for this (mT, lane) column
      if (t + 1 < T) {
        int m0 = mT * 16 + ((lane >> 4) << 3);
        __builtin_prefetch(xz + ((size_t)m0 * T + (t + 1)) * N + n, 0, 1);
      }
      #pragma unroll
      for (int kT = 0; kT < kTiles; ++kT) { // z += h_{t-1} @ U
        const unsigned short* bp = Ulds + ((((nTloc << 4) + kT) * 32 + lane) << 4);
        union { v16bf v; v8bf h[2]; } bu;
        bu.h[0] = *(const v8bf*)(bp);
        bu.h[1] = *(const v8bf*)(bp + 8);
        v16bf a = load_a_row(hb + (size_t)(mT * 16 + (lane & 15)) * K, kT << 5, lane);
        acc = wmma_bf16(a, bu.v, acc);
      }
      #pragma unroll
      for (int v = 0; v < 8; ++v) {        // gates (i,f,g,o interleaved in n)
        float z  = acc[v];
        float zi = __shfl(z, baseLane + 0, 32);
        float zf = __shfl(z, baseLane + 1, 32);
        float zg = __shfl(z, baseLane + 2, 32);
        float zo = __shfl(z, baseLane + 3, 32);
        float cc = sigm(zf) * c[mT][v] + sigm(zi) * tanhf(zg);
        c[mT][v] = cc;
        float h = sigm(zo) * tanhf(cc);
        if ((lane & 3) == 0) {
          int m = mT * 16 + v + ((lane >> 4) << 3);
          unsigned short h16 = f32_bf16(h);
          hw[(size_t)m * 512 + u] = h16;
          if (h1out) h1out[((size_t)m * T + t) * 512 + u] = h16;
        }
      }
    }
    // ---- device-wide step barrier across the 16 workgroups ----
    __threadfence();
    __syncthreads();
    if (tid == 0) {
      __hip_atomic_fetch_add(bar, 1u, __ATOMIC_ACQ_REL, __HIP_MEMORY_SCOPE_AGENT);
      unsigned target = (unsigned)NWG * (unsigned)(t + 1);
      while (__hip_atomic_load(bar, __ATOMIC_ACQUIRE, __HIP_MEMORY_SCOPE_AGENT) < target)
        __builtin_amdgcn_s_sleep(2);
    }
    __syncthreads();
    __builtin_amdgcn_fence(__ATOMIC_ACQUIRE, "agent"); // invalidate L0/L1 -> fresh h
  }
}

// ---- output: out = h2_last(64x512) @ Woutsw + bout -----------------------
__global__ __launch_bounds__(256) void k_out_gemm(
    const unsigned short* __restrict__ hlast,  // [64][512] bf16
    const unsigned short* __restrict__ Bsw,    // swizzled 512x512 bf16
    const float* __restrict__ bias,            // [512]
    float* __restrict__ out)                   // [64][512] f32
{
  const int K = 512, N = 512, kTiles = 16;
  int wave = threadIdx.x >> 5, lane = threadIdx.x & 31;
  int nTile = blockIdx.x * 8 + wave;

  v8f acc[4];
  v8f zero = {};
  #pragma unroll
  for (int mT = 0; mT < 4; ++mT) acc[mT] = zero;

  for (int kT = 0; kT < kTiles; ++kT) {
    v16bf b = load_b_sw(Bsw, kTiles, nTile, kT, lane);
    #pragma unroll
    for (int mT = 0; mT < 4; ++mT) {
      v16bf a = load_a_row(hlast + (size_t)(mT * 16 + (lane & 15)) * K, kT << 5, lane);
      acc[mT] = wmma_bf16(a, b, acc[mT]);
    }
  }
  int nn = (nTile << 4) + (lane & 15);
  float bv = bias[nn];
  #pragma unroll
  for (int mT = 0; mT < 4; ++mT)
    #pragma unroll
    for (int v = 0; v < 8; ++v) {
      int m = mT * 16 + v + ((lane >> 4) << 3);
      out[(size_t)m * N + nn] = acc[mT][v] + bv;
    }
}

// ---- host orchestration --------------------------------------------------
extern "C" void kernel_launch(void* const* d_in, const int* in_sizes, int n_in,
                              void* d_out, int out_size, void* d_ws, size_t ws_size,
                              hipStream_t stream) {
  const int*   source = (const int*)  d_in[0];
  const float* emb    = (const float*)d_in[1];
  const float* W0     = (const float*)d_in[2];
  const float* U0     = (const float*)d_in[3];
  const float* b0     = (const float*)d_in[4];
  const float* W1     = (const float*)d_in[5];
  const float* U1     = (const float*)d_in[6];
  const float* b1     = (const float*)d_in[7];
  const float* Wout   = (const float*)d_in[8];
  const float* bout   = (const float*)d_in[9];

  const int B = 64, T = 1024, D = 512, N4 = 2048, VOCAB = 32000;
  const long long M = (long long)B * T; // 65536 rows

  char* ws = (char*)d_ws;
  size_t off = 0;
  auto take = [&](size_t bytes) -> char* {
    char* p = ws + off;
    off = (off + bytes + 255) & ~(size_t)255;
    return p;
  };
  unsigned short* emb16  = (unsigned short*)take((size_t)VOCAB * D * 2);
  unsigned short* W0sw   = (unsigned short*)take((size_t)D * N4 * 2);
  unsigned short* U0sw   = (unsigned short*)take((size_t)D * N4 * 2);
  unsigned short* W1sw   = (unsigned short*)take((size_t)D * N4 * 2);
  unsigned short* U1sw   = (unsigned short*)take((size_t)D * N4 * 2);
  unsigned short* Woutsw = (unsigned short*)take((size_t)D * D * 2);
  float*          b0p    = (float*)take((size_t)N4 * 4);
  float*          b1p    = (float*)take((size_t)N4 * 4);
  unsigned short* xz     = (unsigned short*)take((size_t)M * N4 * 2);   // 256 MB (reused)
  unsigned short* h1     = (unsigned short*)take((size_t)M * D * 2);    // 64 MB
  unsigned short* hbuf   = (unsigned short*)take((size_t)2 * 64 * 512 * 2);
  unsigned int*   bar    = (unsigned int*)take(256);

  // allow 128KB dynamic LDS for the scan kernel (host-side, capture-safe)
  static bool attrDone = false;
  if (!attrDone) {
    (void)hipFuncSetAttribute((const void*)k_lstm_scan,
                              hipFuncAttributeMaxDynamicSharedMemorySize, 131072);
    attrDone = true;
  }

  // --- prep: bf16 conversion + gate-interleave + WMMA-B swizzle ---
  k_f32_to_bf16<<<4096, 256, 0, stream>>>(emb, emb16, (long long)VOCAB * D);
  k_prep_w<<<2048, 256, 0, stream>>>(W0,   W0sw,   D, N4, 1);
  k_prep_w<<<2048, 256, 0, stream>>>(U0,   U0sw,   D, N4, 1);
  k_prep_w<<<2048, 256, 0, stream>>>(W1,   W1sw,   D, N4, 1);
  k_prep_w<<<2048, 256, 0, stream>>>(Wout, Woutsw, D, D,  0);
  k_prep_w<<<2048, 256, 0, stream>>>(U1,   U1sw,   D, N4, 1);
  k_prep_b<<<8, 256, 0, stream>>>(b0, b0p, N4);
  k_prep_b<<<8, 256, 0, stream>>>(b1, b1p, N4);

  dim3 gGemm((unsigned)(M / 64), 16);

  // --- layer 0 ---
  (void)hipMemsetAsync(hbuf, 0, (size_t)2 * 64 * 512 * 2, stream);
  (void)hipMemsetAsync(bar, 0, 256, stream);
  k_gemm_xz<<<gGemm, 256, 0, stream>>>(source, emb16, W0sw, b0p, xz);
  k_lstm_scan<<<16, 256, 131072, stream>>>(xz, U0sw, hbuf, h1, bar, T);

  // --- layer 1 ---
  (void)hipMemsetAsync(hbuf, 0, (size_t)2 * 64 * 512 * 2, stream);
  (void)hipMemsetAsync(bar, 0, 256, stream);
  k_gemm_xz<<<gGemm, 256, 0, stream>>>(nullptr, h1, W1sw, b1p, xz);
  k_lstm_scan<<<16, 256, 131072, stream>>>(xz, U1sw, hbuf, nullptr, bar, T);

  // --- output projection: final h2 lives in ping slot 0 (T even) ---
  k_out_gemm<<<4, 256, 0, stream>>>(hbuf, Woutsw, bout, (float*)d_out);
}
